// VisionTransformerDetFast_50551765074259
// MI455X (gfx1250) — matrix operand order repaired and checked by reading, not compiled, not run on hardware
//
#include <hip/hip_runtime.h>
#include <math.h>

// ---------------- model constants ----------------
#define DEPTH    12
#define DIM      768
#define HEADS    12
#define HEAD_DIM 64
#define MLPD     3072
#define GRID_T   32      // tokens per side
#define S_TOT    1024    // total tokens
#define WIN      16
#define QKV_N    (3*DIM) // 2304

// ---------------- WMMA types (gfx1250, wave32) ----------------
typedef __bf16 bf16_t;
typedef bf16_t v16bf __attribute__((ext_vector_type(16)));
typedef float  v8f   __attribute__((ext_vector_type(8)));

union FragAB { v16bf v; unsigned int u[8]; };
union FragC  { v8f  v; float f[8]; };

__device__ __forceinline__ unsigned short f2bf(float f) {
  unsigned int u = __float_as_uint(f);
  u += 0x7FFFu + ((u >> 16) & 1u);         // round-to-nearest-even
  return (unsigned short)(u >> 16);
}
__device__ __forceinline__ float bf2f(unsigned short h) {
  return __uint_as_float(((unsigned int)h) << 16);
}

// packed fp32x2 -> bf16x2 (prefer the HW packed convert if clang exposes it)
#if __has_builtin(__builtin_amdgcn_cvt_pk_bf16_f32)
__device__ __forceinline__ unsigned int f2bf2(float a, float b) {
  auto r = __builtin_amdgcn_cvt_pk_bf16_f32(a, b);
  unsigned int u;
  __builtin_memcpy(&u, &r, 4);
  return u;
}
#else
__device__ __forceinline__ unsigned int f2bf2(float a, float b) {
  return (unsigned int)f2bf(a) | ((unsigned int)f2bf(b) << 16);
}
#endif

// async global->LDS staging (CDNA5 ASYNCcnt path), guarded.
// Probe-discovered signature: (int4 as(1)*, int4 as(3)*, imm offset, imm cpol)
#if __has_builtin(__builtin_amdgcn_global_load_async_to_lds_b128) && \
    __has_builtin(__builtin_amdgcn_s_wait_asynccnt)
#define USE_ASYNC_LDS 1
typedef int i4v __attribute__((__vector_size__(4 * sizeof(int))));
typedef __attribute__((address_space(1))) i4v* gp4_t;
typedef __attribute__((address_space(3))) i4v* lp4_t;
#else
#define USE_ASYNC_LDS 0
#endif

// token map: window layers regroup the 32x32 grid into 4 windows of 16x16.
__device__ __forceinline__ int tok_of(int p, int li, int is_win) {
  if (is_win) {
    int wy = p >> 1, wx = p & 1;
    int r = li >> 4, c = li & 15;
    return ((wy << 4) + r) * GRID_T + (wx << 4) + c;
  }
  return li;
}

// ---------------- patch embed: conv16x16/s16 == per-token 768x768 GEMV ------
__global__ __launch_bounds__(256)
void k_patch(const float* __restrict__ x, const float* __restrict__ cw,
             const float* __restrict__ cb, const float* __restrict__ pos,
             float* __restrict__ h) {
  __shared__ float xs[768];
  int t = blockIdx.x;
  int gy = t >> 5, gx = t & 31;
  for (int e = threadIdx.x; e < 768; e += 256) {
    int ci = e >> 8, rem = e & 255, py = rem >> 4, px = rem & 15;
    xs[e] = x[((size_t)ci * 512 + gy * 16 + py) * 512 + gx * 16 + px];
  }
  __syncthreads();
  for (int c = threadIdx.x; c < 768; c += 256) {
    const float* wr = cw + (size_t)c * 768;
    float acc = 0.f;
    #pragma unroll 8
    for (int e = 0; e < 768; ++e) acc += xs[e] * wr[e];
    h[(size_t)t * 768 + c] = acc + cb[c] + pos[(size_t)t * 768 + c];
  }
}

// ---------------- layernorm -> bf16 activations ----------------
__global__ __launch_bounds__(256)
void k_ln(const float* __restrict__ x, const float* __restrict__ s,
          const float* __restrict__ b, unsigned short* __restrict__ out) {
  __shared__ float r1[256], r2[256];
  int t = blockIdx.x, tid = threadIdx.x;
  const float* row = x + (size_t)t * 768;
  float v0 = row[tid], v1 = row[tid + 256], v2 = row[tid + 512];
  r1[tid] = v0 + v1 + v2;
  r2[tid] = v0 * v0 + v1 * v1 + v2 * v2;
  __syncthreads();
  for (int st = 128; st > 0; st >>= 1) {
    if (tid < st) { r1[tid] += r1[tid + st]; r2[tid] += r2[tid + st]; }
    __syncthreads();
  }
  float m   = r1[0] * (1.f / 768.f);
  float var = r2[0] * (1.f / 768.f) - m * m;
  float inv = rsqrtf(var + 1e-5f);
  unsigned short* o = out + (size_t)t * 768;
  o[tid]       = f2bf((v0 - m) * inv * s[tid]       + b[tid]);
  o[tid + 256] = f2bf((v1 - m) * inv * s[tid + 256] + b[tid + 256]);
  o[tid + 512] = f2bf((v2 - m) * inv * s[tid + 512] + b[tid + 512]);
}

// ---------------- bf16 WMMA GEMM ----------------
// C[M,N] = act(A_bf16[M,K] @ W_f32[K,N] + bias) (+res) -> f32 and/or bf16.
// Block tile 128x128, K-step 64 (16 WMMAs per barrier pair), 8 waves,
// each wave 32(M)x64(N) = 2x4 wmma tiles. W converted fp32->bf16 while
// staging into LDS (weight reuse across M comes from the 192MB L2).
__global__ __launch_bounds__(256)
void k_gemm(const unsigned short* __restrict__ A, const float* __restrict__ W,
            const float* __restrict__ bias, const float* __restrict__ res,
            float* __restrict__ outF, unsigned short* __restrict__ outB,
            int M, int N, int K, int gelu) {
  __shared__ unsigned short As[128][64];   // [m][k] bf16, 16KB
  __shared__ unsigned short Bs[128][64];   // [n][k] bf16 (transposed), 16KB
  int tid  = threadIdx.x;
  int lane = tid & 31, wid = tid >> 5;
  int wm = wid >> 1, wn = wid & 1;
  int l15 = lane & 15, hi = lane >> 4;
  int m0 = blockIdx.y * 128, n0 = blockIdx.x * 128;

  FragC acc[2][4];
  #pragma unroll
  for (int i = 0; i < 2; ++i)
    #pragma unroll
    for (int j = 0; j < 4; ++j)
      #pragma unroll
      for (int r = 0; r < 8; ++r) acc[i][j].f[r] = 0.f;

  int ar = tid >> 1, ah = tid & 1;         // A stage: 64B per thread
  const unsigned short* gA = A + (size_t)(m0 + ar) * K + ah * 32;
  int bk = tid >> 2, bseg = tid & 3;       // B stage: 32 f32 per thread
  const float* gBrow = W + (size_t)bk * N + n0 + bseg * 32;

  for (int k0 = 0; k0 < K; k0 += 64) {
    // ---- stage A tile (bf16 copy; async DMA to LDS when available) ----
#if USE_ASYNC_LDS
    {
      gp4_t gsrc = (gp4_t)(gA + k0);
      lp4_t ldst = (lp4_t)&As[ar][ah * 32];
      __builtin_amdgcn_global_load_async_to_lds_b128(gsrc, ldst, 0, 0);
      __builtin_amdgcn_global_load_async_to_lds_b128(gsrc, ldst, 16, 0);
      __builtin_amdgcn_global_load_async_to_lds_b128(gsrc, ldst, 32, 0);
      __builtin_amdgcn_global_load_async_to_lds_b128(gsrc, ldst, 48, 0);
    }
#else
    {
      const uint4* pA = (const uint4*)(gA + k0);
      uint4 a0 = pA[0], a1 = pA[1], a2 = pA[2], a3 = pA[3];
      *(uint4*)&As[ar][ah * 32]      = a0;
      *(uint4*)&As[ar][ah * 32 + 8]  = a1;
      *(uint4*)&As[ar][ah * 32 + 16] = a2;
      *(uint4*)&As[ar][ah * 32 + 24] = a3;
    }
#endif
    // ---- stage B tile (fp32 -> bf16 packed convert, transpose to [n][k]) --
    const float* gB = gBrow + (size_t)k0 * N;
    unsigned int tmp2[16];
    #pragma unroll
    for (int j = 0; j < 16; ++j) tmp2[j] = f2bf2(gB[2 * j], gB[2 * j + 1]);
    const unsigned short* ts = (const unsigned short*)tmp2;
    #pragma unroll
    for (int j = 0; j < 32; ++j) Bs[bseg * 32 + j][bk] = ts[j];
    if (k0 + 64 < K) {
      __builtin_prefetch(gA + k0 + 64, 0, 1);
      __builtin_prefetch(gB + (size_t)64 * N, 0, 1);
    }
#if USE_ASYNC_LDS
    __builtin_amdgcn_s_wait_asynccnt(0);
#endif
    __syncthreads();

    #pragma unroll
    for (int s = 0; s < 2; ++s) {
      FragAB fa[2], fb[4];
      #pragma unroll
      for (int i = 0; i < 2; ++i) {
        int row = wm * 32 + i * 16 + l15;
        const unsigned int* p0 = (const unsigned int*)&As[row][s * 32 + hi * 16];
        const unsigned int* p1 = (const unsigned int*)&As[row][s * 32 + hi * 16 + 8];
        fa[i].u[0] = p0[0]; fa[i].u[1] = p0[1]; fa[i].u[2] = p0[2]; fa[i].u[3] = p0[3];
        fa[i].u[4] = p1[0]; fa[i].u[5] = p1[1]; fa[i].u[6] = p1[2]; fa[i].u[7] = p1[3];
      }
      #pragma unroll
      for (int j = 0; j < 4; ++j) {
        int col = wn * 64 + j * 16 + l15;
        const unsigned int* p = (const unsigned int*)&Bs[col][s * 32 + hi * 16];
        #pragma unroll
        for (int q = 0; q < 8; ++q) fb[j].u[q] = p[q];
      }
      #pragma unroll
      for (int i = 0; i < 2; ++i)
        #pragma unroll
        for (int j = 0; j < 4; ++j)
          acc[i][j].v = __builtin_amdgcn_wmma_f32_16x16x32_bf16(
              false, fa[i].v, false, fb[j].v, (short)0, acc[i][j].v, false, false);
    }
    __syncthreads();
  }

  // epilogue: bias (+ exact GELU) (+ residual) -> f32 and/or bf16 outputs
  #pragma unroll
  for (int i = 0; i < 2; ++i) {
    #pragma unroll
    for (int j = 0; j < 4; ++j) {
      int col = n0 + wn * 64 + j * 16 + l15;
      float bv = bias ? bias[col] : 0.f;
      #pragma unroll
      for (int r = 0; r < 8; ++r) {
        int row = m0 + wm * 32 + i * 16 + r + hi * 8;
        float v = acc[i][j].f[r] + bv;
        if (gelu) v = 0.5f * v * (1.f + erff(v * 0.70710678118f));
        size_t off = (size_t)row * N + col;
        if (res)  v += res[off];
        if (outF) outF[off] = v;
        if (outB) outB[off] = f2bf(v);
      }
    }
  }
}

// ---------------- decomposed rel-pos bias: rel[n][tok][k] = q . R[idx] ------
__global__ __launch_bounds__(256)
void k_relpos(const unsigned short* __restrict__ qkv, const float* __restrict__ R,
              float* __restrict__ rel, int Wk, int is_win, int use_x) {
  int idx = blockIdx.x * 256 + threadIdx.x;
  int total = HEADS * S_TOT * Wk;
  if (idx >= total) return;
  int kk = idx % Wk;
  int t  = (idx / Wk) % S_TOT;
  int n  = idx / (Wk * S_TOT);
  int gy = t >> 5, gx = t & 31;
  int coord = use_x ? gx : gy;
  if (is_win) coord &= 15;
  int ridx = coord - kk + Wk - 1;
  const unsigned short* q = qkv + (size_t)t * QKV_N + n * HEAD_DIM;
  const float* Rr = R + (size_t)ridx * HEAD_DIM;
  float s = 0.f;
  #pragma unroll
  for (int c = 0; c < HEAD_DIM; ++c) s += bf2f(q[c]) * Rr[c];
  rel[((size_t)n * S_TOT + t) * 32 + kk] = s;
}

// ---------------- attention logits: 16x16 tile per wave, 2 WMMAs (d=64) -----
__global__ __launch_bounds__(256)
void k_scores(const unsigned short* __restrict__ qkv,
              const float* __restrict__ relh, const float* __restrict__ relw,
              float* __restrict__ scores, int P, int S, int is_win) {
  int tid = threadIdx.x, lane = tid & 31, wid = tid >> 5;
  int l15 = lane & 15, hi = lane >> 4;
  int n = blockIdx.z / P, p = blockIdx.z % P;
  int qt = blockIdx.y, kt = blockIdx.x * 8 + wid;

  int qtok = tok_of(p, qt * 16 + l15, is_win);
  int ktok = tok_of(p, kt * 16 + l15, is_win);
  const unsigned short* qb = qkv + (size_t)qtok * QKV_N + n * HEAD_DIM;
  const unsigned short* kb = qkv + (size_t)ktok * QKV_N + DIM + n * HEAD_DIM;

  FragC acc;
  #pragma unroll
  for (int r = 0; r < 8; ++r) acc.f[r] = 0.f;
  #pragma unroll
  for (int st = 0; st < 2; ++st) {
    FragAB fa, fb;
    const unsigned short* qp = qb + st * 32 + hi * 16;
    uint4 c0 = *(const uint4*)qp;
    uint4 c1 = *(const uint4*)(qp + 8);
    *(uint4*)&fa.u[0] = c0; *(uint4*)&fa.u[4] = c1;
    const unsigned short* kp = kb + st * 32 + hi * 16;
    uint4 b0 = *(const uint4*)kp;
    uint4 b1 = *(const uint4*)(kp + 8);
    *(uint4*)&fb.u[0] = b0; *(uint4*)&fb.u[4] = b1;
    acc.v = __builtin_amdgcn_wmma_f32_16x16x32_bf16(
        false, fa.v, false, fb.v, (short)0, acc.v, false, false);
  }

  int kl = kt * 16 + l15;
  int kr = is_win ? (kl >> 4) : (kl >> 5);
  int kc = is_win ? (kl & 15) : (kl & 31);
  const float scale = 0.125f;   // 1/sqrt(64)
  #pragma unroll
  for (int r = 0; r < 8; ++r) {
    int ql = qt * 16 + r + hi * 8;
    int qt_tok = tok_of(p, ql, is_win);
    size_t rb = ((size_t)n * S_TOT + qt_tok) * 32;
    float v = acc.f[r] * scale + relh[rb + kr] + relw[rb + kc];
    scores[(((size_t)(n * P + p)) * S + ql) * S + kl] = v;
  }
}

// ---------------- row softmax -> bf16 probs ----------------
__global__ __launch_bounds__(256)
void k_softmax(const float* __restrict__ scores, unsigned short* __restrict__ probs,
               int Sk) {
  __shared__ float red[256];
  int tid = threadIdx.x;
  size_t base = ((size_t)blockIdx.y * gridDim.x + blockIdx.x) * Sk;
  int cnt = (Sk + 255) >> 8;
  float vals[4];
  float mx = -3.4e38f;
  for (int c = 0; c < cnt; ++c) {
    int i = tid + c * 256;
    vals[c] = (i < Sk) ? scores[base + i] : -3.4e38f;
    mx = fmaxf(mx, vals[c]);
  }
  red[tid] = mx; __syncthreads();
  for (int st = 128; st > 0; st >>= 1) {
    if (tid < st) red[tid] = fmaxf(red[tid], red[tid + st]);
    __syncthreads();
  }
  mx = red[0]; __syncthreads();
  float sum = 0.f;
  for (int c = 0; c < cnt; ++c) {
    int i = tid + c * 256;
    if (i < Sk) { vals[c] = __expf(vals[c] - mx); sum += vals[c]; }
  }
  red[tid] = sum; __syncthreads();
  for (int st = 128; st > 0; st >>= 1) {
    if (tid < st) red[tid] += red[tid + st];
    __syncthreads();
  }
  float inv = 1.f / red[0];
  for (int c = 0; c < cnt; ++c) {
    int i = tid + c * 256;
    if (i < Sk) probs[base + i] = f2bf(vals[c] * inv);
  }
}

// ---------------- out = probs @ V : 128 q-rows x 64 dims per block ----------
__global__ __launch_bounds__(256)
void k_pv(const unsigned short* __restrict__ probs,
          const unsigned short* __restrict__ qkv,
          unsigned short* __restrict__ attnb, int P, int S, int is_win) {
  __shared__ unsigned short Vt[64][32];    // [d][k] bf16
  int tid = threadIdx.x, lane = tid & 31, wid = tid >> 5;
  int l15 = lane & 15, hi = lane >> 4;
  int n = blockIdx.y / P, p = blockIdx.y % P;
  int q0 = blockIdx.x * 128;

  FragC acc[4];
  #pragma unroll
  for (int j = 0; j < 4; ++j)
    #pragma unroll
    for (int r = 0; r < 8; ++r) acc[j].f[r] = 0.f;

  int vkey = tid >> 3, vdseg = (tid & 7) * 8;
  size_t prow = ((size_t)(n * P + p) * S + q0 + wid * 16 + l15) * S;

  for (int kk = 0; kk < S; kk += 32) {
    // stage V tile transposed into LDS
    int vtok = tok_of(p, kk + vkey, is_win);
    const unsigned short* vp = qkv + (size_t)vtok * QKV_N + 2 * DIM + n * HEAD_DIM + vdseg;
    uint4 vv = *(const uint4*)vp;
    unsigned short vs[8];
    *(uint4*)vs = vv;
    #pragma unroll
    for (int j = 0; j < 8; ++j) Vt[vdseg + j][vkey] = vs[j];
    __syncthreads();

    FragAB fa, fb[4];
    const unsigned short* ap = probs + prow + kk + hi * 16;
    uint4 c0 = *(const uint4*)ap;
    uint4 c1 = *(const uint4*)(ap + 8);
    *(uint4*)&fa.u[0] = c0; *(uint4*)&fa.u[4] = c1;
    #pragma unroll
    for (int j = 0; j < 4; ++j) {
      int col = j * 16 + l15;
      const unsigned int* bp = (const unsigned int*)&Vt[col][hi * 16];
      #pragma unroll
      for (int q = 0; q < 8; ++q) fb[j].u[q] = bp[q];
    }
    #pragma unroll
    for (int j = 0; j < 4; ++j)
      acc[j].v = __builtin_amdgcn_wmma_f32_16x16x32_bf16(
          false, fa.v, false, fb[j].v, (short)0, acc[j].v, false, false);
    __syncthreads();
  }

  #pragma unroll
  for (int j = 0; j < 4; ++j) {
    #pragma unroll
    for (int r = 0; r < 8; ++r) {
      int ql = q0 + wid * 16 + r + hi * 8;
      int tok = tok_of(p, ql, is_win);
      attnb[(size_t)tok * DIM + n * HEAD_DIM + j * 16 + l15] = f2bf(acc[j].f[r]);
    }
  }
}

// ---------------- driver ----------------
extern "C" void kernel_launch(void* const* d_in, const int* in_sizes, int n_in,
                              void* d_out, int out_size, void* d_ws, size_t ws_size,
                              hipStream_t stream) {
  const float* x        = (const float*)d_in[0];
  const float* conv_w   = (const float*)d_in[1];
  const float* conv_b   = (const float*)d_in[2];
  const float* pos      = (const float*)d_in[3];
  const float* ln1_s    = (const float*)d_in[4];
  const float* ln1_b    = (const float*)d_in[5];
  const float* qkv_w    = (const float*)d_in[6];
  const float* qkv_b    = (const float*)d_in[7];
  const float* proj_w   = (const float*)d_in[8];
  const float* proj_b   = (const float*)d_in[9];
  const float* rh_win   = (const float*)d_in[10];
  const float* rw_win   = (const float*)d_in[11];
  const float* rh_glob  = (const float*)d_in[12];
  const float* rw_glob  = (const float*)d_in[13];
  const float* ln2_s    = (const float*)d_in[14];
  const float* ln2_b    = (const float*)d_in[15];
  const float* fc1_w    = (const float*)d_in[16];
  const float* fc1_b    = (const float*)d_in[17];
  const float* fc2_w    = (const float*)d_in[18];
  const float* fc2_b    = (const float*)d_in[19];

  unsigned char* wsp = (unsigned char*)d_ws;
  auto take = [&](size_t bytes) -> void* {
    void* pv = (void*)wsp;
    wsp += (bytes + 255) & ~(size_t)255;
    return pv;
  };
  float*          hb    = (float*)take((size_t)S_TOT * DIM * 4);
  unsigned short* xln   = (unsigned short*)take((size_t)S_TOT * DIM * 2);
  unsigned short* qkvb  = (unsigned short*)take((size_t)S_TOT * QKV_N * 2);
  unsigned short* attnb = (unsigned short*)take((size_t)S_TOT * DIM * 2);
  unsigned short* midb  = (unsigned short*)take((size_t)S_TOT * MLPD * 2);
  float*          relh  = (float*)take((size_t)HEADS * S_TOT * 32 * 4);
  float*          relw  = (float*)take((size_t)HEADS * S_TOT * 32 * 4);
  float*          scbuf = (float*)take((size_t)HEADS * S_TOT * S_TOT * 4);
  unsigned short* prbuf = (unsigned short*)take((size_t)HEADS * S_TOT * S_TOT * 2);

  k_patch<<<S_TOT, 256, 0, stream>>>(x, conv_w, conv_b, pos, hb);

  int wi = 0, gi = 0;
  for (int i = 0; i < DEPTH; ++i) {
    const bool g = (i == 2 || i == 5 || i == 8 || i == 11);
    const int P = g ? 1 : 4;
    const int S = g ? S_TOT : 256;
    const int Wk = g ? GRID_T : WIN;
    const float* Rh = g ? rh_glob + (size_t)gi * (2 * GRID_T - 1) * HEAD_DIM
                        : rh_win  + (size_t)wi * (2 * WIN - 1) * HEAD_DIM;
    const float* Rw = g ? rw_glob + (size_t)gi * (2 * GRID_T - 1) * HEAD_DIM
                        : rw_win  + (size_t)wi * (2 * WIN - 1) * HEAD_DIM;

    // attn branch
    k_ln<<<S_TOT, 256, 0, stream>>>(hb, ln1_s + i * DIM, ln1_b + i * DIM, xln);
    k_gemm<<<dim3(QKV_N / 128, S_TOT / 128), 256, 0, stream>>>(
        xln, qkv_w + (size_t)i * DIM * QKV_N, qkv_b + i * QKV_N,
        nullptr, nullptr, qkvb, S_TOT, QKV_N, DIM, 0);
    int rt = HEADS * S_TOT * Wk;
    k_relpos<<<(rt + 255) / 256, 256, 0, stream>>>(qkvb, Rh, relh, Wk, !g, 0);
    k_relpos<<<(rt + 255) / 256, 256, 0, stream>>>(qkvb, Rw, relw, Wk, !g, 1);
    k_scores<<<dim3(S / 128, S / 16, HEADS * P), 256, 0, stream>>>(
        qkvb, relh, relw, scbuf, P, S, !g);
    k_softmax<<<dim3(S, HEADS * P), 256, 0, stream>>>(scbuf, prbuf, S);
    k_pv<<<dim3(S / 128, HEADS * P), 256, 0, stream>>>(prbuf, qkvb, attnb, P, S, !g);
    // proj + fused residual (h = h + attn_proj)
    k_gemm<<<dim3(DIM / 128, S_TOT / 128), 256, 0, stream>>>(
        attnb, proj_w + (size_t)i * DIM * DIM, proj_b + i * DIM,
        hb, hb, nullptr, S_TOT, DIM, DIM, 0);

    // mlp branch
    k_ln<<<S_TOT, 256, 0, stream>>>(hb, ln2_s + i * DIM, ln2_b + i * DIM, xln);
    k_gemm<<<dim3(MLPD / 128, S_TOT / 128), 256, 0, stream>>>(
        xln, fc1_w + (size_t)i * DIM * MLPD, fc1_b + i * MLPD,
        nullptr, nullptr, midb, S_TOT, MLPD, DIM, 1);
    // fc2 + fused residual; last layer streams straight into d_out
    float* dst = (i == DEPTH - 1) ? (float*)d_out : hb;
    k_gemm<<<dim3(DIM / 128, S_TOT / 128), 256, 0, stream>>>(
        midb, fc2_w + (size_t)i * MLPD * DIM, fc2_b + i * DIM,
        hb, dst, nullptr, S_TOT, DIM, MLPD, 0);

    if (g) ++gi; else ++wi;
  }
}
